// MeshGraphNetProcessor_88510686036702
// MI455X (gfx1250) — compile-verified, matrix operand order checked by reading, not run on hardware
//
#include <hip/hip_runtime.h>

#define D        128
#define NN       40000
#define NE       640000
#define NLAYERS  15
#define LN_EPS   1e-5f

typedef __attribute__((ext_vector_type(16))) __bf16 v16bf;
typedef __attribute__((ext_vector_type(8)))  __bf16 v8bf;
typedef __attribute__((ext_vector_type(8)))  float  v8f;

// ---- bf16 helpers -------------------------------------------------------
__device__ __forceinline__ unsigned short f2bf_bits(float f) {
    union { __bf16 h; unsigned short u; } c;
    c.h = (__bf16)f;
    return c.u;
}

// 32-byte fragment load via two 16-byte loads (only 16B alignment guaranteed in LDS)
__device__ __forceinline__ v16bf load_frag(const __bf16* p) {
    v8bf lo = *(const v8bf*)p;
    v8bf hi = *(const v8bf*)(p + 8);
    return __builtin_shufflevector(lo, hi, 0,1,2,3,4,5,6,7,8,9,10,11,12,13,14,15);
}

__device__ __forceinline__ v8f wmma_bf16(v16bf a, v16bf b, v8f c) {
    return __builtin_amdgcn_wmma_f32_16x16x32_bf16(
        /*neg_a=*/false, a, /*neg_b=*/false, b,
        /*c_mod=*/(short)0, c, /*reuse_a=*/false, /*reuse_b=*/false);
}

// ---- weight convert: W (din x dout f32, row-major) -> Wt (dout x din bf16) ----
__global__ void convert_wt(const float* __restrict__ W, unsigned short* __restrict__ Wt,
                           int din, int dout) {
    int i = blockIdx.x * blockDim.x + threadIdx.x;
    if (i >= din * dout) return;
    int o = i / din, in_ = i - o * din;
    Wt[i] = f2bf_bits(W[(size_t)in_ * dout + o]);
}

// ---- edge block: e += LN(MLP([n[src], n[dst], e])) ----------------------
__launch_bounds__(256)
__global__ void edge_block(
    const float* __restrict__ nfeat,
    float* __restrict__ e,                       // updated in place (disjoint rows per WG)
    const int* __restrict__ eidx,                // [2][NE]
    const unsigned short* __restrict__ W1u,      // [128][384] bf16
    const unsigned short* __restrict__ W2u,      // [128][128]
    const unsigned short* __restrict__ W3u,      // [128][128]
    const float* __restrict__ b1, const float* __restrict__ b2,
    const float* __restrict__ b3,
    const float* __restrict__ gw, const float* __restrict__ gb)
{
    constexpr int SX = 384 + 8;                  // padded strides (bank-conflict free)
    constexpr int SH = 128 + 8;
    __shared__ __align__(16) __bf16 Xs[128 * SX];   // 100352 B (reused as h2, stride SH)
    __shared__ __align__(16) __bf16 Hs[128 * SH];   // 34816 B
    __shared__ int sIdx[128], dIdx[128];

    const __bf16* W1t = (const __bf16*)W1u;
    const __bf16* W2t = (const __bf16*)W2u;
    const __bf16* W3t = (const __bf16*)W3u;

    const int tid  = threadIdx.x;
    const int lane = tid & 31;
    const int wave = tid >> 5;
    const int l    = lane & 15;
    const int half = lane >> 4;
    const int ebase = blockIdx.x * 128;

    if (tid < 128) {
        sIdx[tid] = eidx[ebase + tid];
        dIdx[tid] = eidx[NE + ebase + tid];
    }
    __syncthreads();

    // Build X = [n[src] | n[dst] | e] as bf16
    for (int it = tid; it < 128 * 96; it += 256) {
        int row = it / 96;
        int c4  = (it - row * 96) * 4;
        const float* sp;
        if (c4 < 128)      sp = nfeat + (size_t)sIdx[row] * D + c4;
        else if (c4 < 256) sp = nfeat + (size_t)dIdx[row] * D + (c4 - 128);
        else               sp = e + (size_t)(ebase + row) * D + (c4 - 256);
        float4 v = *(const float4*)sp;
        __bf16* dp = &Xs[row * SX + c4];
        dp[0] = (__bf16)v.x; dp[1] = (__bf16)v.y;
        dp[2] = (__bf16)v.z; dp[3] = (__bf16)v.w;
    }
    __syncthreads();

    const int mrowA = wave * 16 + l;
    const int mst   = wave * 16 + half * 8;

    // GEMM1: h1 = relu(X @ W1 + b1)  -> Hs
    {
        const __bf16* aB = &Xs[mrowA * SX + half * 16];
        for (int t = 0; t < 8; ++t) {
            const __bf16* bB = W1t + ((t * 16 + l) * 384 + half * 16);
            v8f acc = {};
#pragma unroll
            for (int k = 0; k < 384; k += 32)
                acc = wmma_bf16(load_frag(aB + k), load_frag(bB + k), acc);
            float bias = b1[t * 16 + l];
#pragma unroll
            for (int r = 0; r < 8; ++r) {
                float v = acc[r] + bias;
                Hs[(mst + r) * SH + t * 16 + l] = (__bf16)(v > 0.f ? v : 0.f);
            }
        }
    }
    __syncthreads();

    // GEMM2: h2 = relu(h1 @ W2 + b2) -> Xs region (stride SH)
    {
        const __bf16* aB = &Hs[mrowA * SH + half * 16];
        for (int t = 0; t < 8; ++t) {
            const __bf16* bB = W2t + ((t * 16 + l) * 128 + half * 16);
            v8f acc = {};
#pragma unroll
            for (int k = 0; k < 128; k += 32)
                acc = wmma_bf16(load_frag(aB + k), load_frag(bB + k), acc);
            float bias = b2[t * 16 + l];
#pragma unroll
            for (int r = 0; r < 8; ++r) {
                float v = acc[r] + bias;
                Xs[(mst + r) * SH + t * 16 + l] = (__bf16)(v > 0.f ? v : 0.f);
            }
        }
    }
    __syncthreads();

    // GEMM3 + bias + LayerNorm + residual
    {
        const __bf16* aB = &Xs[mrowA * SH + half * 16];
        v8f acc3[8];
        for (int t = 0; t < 8; ++t) {
            const __bf16* bB = W3t + ((t * 16 + l) * 128 + half * 16);
            v8f acc = {};
#pragma unroll
            for (int k = 0; k < 128; k += 32)
                acc = wmma_bf16(load_frag(aB + k), load_frag(bB + k), acc);
            float bias = b3[t * 16 + l];
#pragma unroll
            for (int r = 0; r < 8; ++r) acc[r] += bias;
            acc3[t] = acc;
        }
        float s[8], q[8];
#pragma unroll
        for (int r = 0; r < 8; ++r) { s[r] = 0.f; q[r] = 0.f; }
#pragma unroll
        for (int t = 0; t < 8; ++t)
#pragma unroll
            for (int r = 0; r < 8; ++r) { float v = acc3[t][r]; s[r] += v; q[r] += v * v; }
#pragma unroll
        for (int m = 1; m < 16; m <<= 1)
#pragma unroll
            for (int r = 0; r < 8; ++r) {
                s[r] += __shfl_xor(s[r], m, 32);
                q[r] += __shfl_xor(q[r], m, 32);
            }
        float mu[8], rstd[8];
#pragma unroll
        for (int r = 0; r < 8; ++r) {
            mu[r] = s[r] * (1.f / 128.f);
            rstd[r] = rsqrtf(q[r] * (1.f / 128.f) - mu[r] * mu[r] + LN_EPS);
        }
        for (int t = 0; t < 8; ++t) {
            float gv = gw[t * 16 + l];
            float bv = gb[t * 16 + l];
#pragma unroll
            for (int r = 0; r < 8; ++r) {
                size_t gi = (size_t)(ebase + mst + r) * D + t * 16 + l;
                e[gi] = e[gi] + (acc3[t][r] - mu[r]) * rstd[r] * gv + bv;
            }
        }
    }
}

// ---- scatter: agg[dst] += e ---------------------------------------------
__global__ void scatter_add(const float* __restrict__ e, const int* __restrict__ eidx,
                            float* __restrict__ agg) {
    int t = blockIdx.x * blockDim.x + threadIdx.x;     // NE*32 threads, 4 floats each
    int edge = t >> 5;
    int c4 = (t & 31) * 4;
    int d = eidx[NE + edge];
    const float* ep = e + (size_t)edge * D + c4;
    float* ap = agg + (size_t)d * D + c4;
    atomicAdd(ap + 0, ep[0]);
    atomicAdd(ap + 1, ep[1]);
    atomicAdd(ap + 2, ep[2]);
    atomicAdd(ap + 3, ep[3]);
}

// ---- node block: n += LN(MLP([n, agg])) ---------------------------------
__launch_bounds__(256)
__global__ void node_block(
    float* __restrict__ n,                         // in place (disjoint rows per WG)
    const float* __restrict__ agg,
    const unsigned short* __restrict__ W1u,        // [128][256] bf16
    const unsigned short* __restrict__ W2u,
    const unsigned short* __restrict__ W3u,
    const float* __restrict__ b1, const float* __restrict__ b2,
    const float* __restrict__ b3,
    const float* __restrict__ gw, const float* __restrict__ gb)
{
    constexpr int SX = 256 + 8;
    constexpr int SH = 128 + 8;
    __shared__ __align__(16) __bf16 Xs[128 * SX];   // 67584 B (reused as h2, stride SH)
    __shared__ __align__(16) __bf16 Hs[128 * SH];   // 34816 B

    const __bf16* W1t = (const __bf16*)W1u;
    const __bf16* W2t = (const __bf16*)W2u;
    const __bf16* W3t = (const __bf16*)W3u;

    const int tid  = threadIdx.x;
    const int lane = tid & 31;
    const int wave = tid >> 5;
    const int l    = lane & 15;
    const int half = lane >> 4;
    const int nbase = blockIdx.x * 128;

    for (int it = tid; it < 128 * 64; it += 256) {
        int row = it / 64;
        int c4  = (it - row * 64) * 4;
        int gr = nbase + row; if (gr >= NN) gr = NN - 1;
        const float* sp = (c4 < 128) ? n + (size_t)gr * D + c4
                                     : agg + (size_t)gr * D + (c4 - 128);
        float4 v = *(const float4*)sp;
        __bf16* dp = &Xs[row * SX + c4];
        dp[0] = (__bf16)v.x; dp[1] = (__bf16)v.y;
        dp[2] = (__bf16)v.z; dp[3] = (__bf16)v.w;
    }
    __syncthreads();

    const int mrowA = wave * 16 + l;
    const int mst   = wave * 16 + half * 8;

    { // GEMM1
        const __bf16* aB = &Xs[mrowA * SX + half * 16];
        for (int t = 0; t < 8; ++t) {
            const __bf16* bB = W1t + ((t * 16 + l) * 256 + half * 16);
            v8f acc = {};
#pragma unroll
            for (int k = 0; k < 256; k += 32)
                acc = wmma_bf16(load_frag(aB + k), load_frag(bB + k), acc);
            float bias = b1[t * 16 + l];
#pragma unroll
            for (int r = 0; r < 8; ++r) {
                float v = acc[r] + bias;
                Hs[(mst + r) * SH + t * 16 + l] = (__bf16)(v > 0.f ? v : 0.f);
            }
        }
    }
    __syncthreads();
    { // GEMM2
        const __bf16* aB = &Hs[mrowA * SH + half * 16];
        for (int t = 0; t < 8; ++t) {
            const __bf16* bB = W2t + ((t * 16 + l) * 128 + half * 16);
            v8f acc = {};
#pragma unroll
            for (int k = 0; k < 128; k += 32)
                acc = wmma_bf16(load_frag(aB + k), load_frag(bB + k), acc);
            float bias = b2[t * 16 + l];
#pragma unroll
            for (int r = 0; r < 8; ++r) {
                float v = acc[r] + bias;
                Xs[(mst + r) * SH + t * 16 + l] = (__bf16)(v > 0.f ? v : 0.f);
            }
        }
    }
    __syncthreads();
    { // GEMM3 + LN + residual (masked tail)
        const __bf16* aB = &Xs[mrowA * SH + half * 16];
        v8f acc3[8];
        for (int t = 0; t < 8; ++t) {
            const __bf16* bB = W3t + ((t * 16 + l) * 128 + half * 16);
            v8f acc = {};
#pragma unroll
            for (int k = 0; k < 128; k += 32)
                acc = wmma_bf16(load_frag(aB + k), load_frag(bB + k), acc);
            float bias = b3[t * 16 + l];
#pragma unroll
            for (int r = 0; r < 8; ++r) acc[r] += bias;
            acc3[t] = acc;
        }
        float s[8], q[8];
#pragma unroll
        for (int r = 0; r < 8; ++r) { s[r] = 0.f; q[r] = 0.f; }
#pragma unroll
        for (int t = 0; t < 8; ++t)
#pragma unroll
            for (int r = 0; r < 8; ++r) { float v = acc3[t][r]; s[r] += v; q[r] += v * v; }
#pragma unroll
        for (int m = 1; m < 16; m <<= 1)
#pragma unroll
            for (int r = 0; r < 8; ++r) {
                s[r] += __shfl_xor(s[r], m, 32);
                q[r] += __shfl_xor(q[r], m, 32);
            }
        float mu[8], rstd[8];
#pragma unroll
        for (int r = 0; r < 8; ++r) {
            mu[r] = s[r] * (1.f / 128.f);
            rstd[r] = rsqrtf(q[r] * (1.f / 128.f) - mu[r] * mu[r] + LN_EPS);
        }
        for (int t = 0; t < 8; ++t) {
            float gv = gw[t * 16 + l];
            float bv = gb[t * 16 + l];
#pragma unroll
            for (int r = 0; r < 8; ++r) {
                int grow = nbase + mst + r;
                if (grow < NN) {
                    size_t gi = (size_t)grow * D + t * 16 + l;
                    n[gi] = n[gi] + (acc3[t][r] - mu[r]) * rstd[r] * gv + bv;
                }
            }
        }
    }
}

// ---- host orchestration --------------------------------------------------
// bf16 weight layout per layer (element offsets):
//   edge W1t 0..49151, W2t 49152.., W3t 65536.., node W1t 81920.., W2t 114688.., W3t 131072..
#define LW_EW1 0
#define LW_EW2 49152
#define LW_EW3 65536
#define LW_NW1 81920
#define LW_NW2 114688
#define LW_NW3 131072
#define LW_SZ  147456

extern "C" void kernel_launch(void* const* d_in, const int* in_sizes, int n_in,
                              void* d_out, int out_size, void* d_ws, size_t ws_size,
                              hipStream_t stream) {
    (void)in_sizes; (void)n_in; (void)out_size; (void)ws_size;

    const float* nf   = (const float*)d_in[0];
    const float* ef   = (const float*)d_in[1];
    const int*   eidx = (const int*)d_in[2];

    char* w = (char*)d_ws;
    float* nA  = (float*)w;  w += (size_t)NN * D * sizeof(float);
    float* agg = (float*)w;  w += (size_t)NN * D * sizeof(float);
    float* eA  = (float*)w;  w += (size_t)NE * D * sizeof(float);
    unsigned short* wt = (unsigned short*)w;   // 15 * LW_SZ bf16 elems

    hipMemcpyAsync(nA, nf, (size_t)NN * D * sizeof(float), hipMemcpyDeviceToDevice, stream);
    hipMemcpyAsync(eA, ef, (size_t)NE * D * sizeof(float), hipMemcpyDeviceToDevice, stream);

    // Leaf order per layer (jax pytree, keys sorted):
    //   edge: W1,W2,W3,b1,b2,b3,beta,g   then node: same
    for (int L = 0; L < NLAYERS; ++L) {
        int base = 3 + L * 16;
        unsigned short* lw = wt + (size_t)L * LW_SZ;
        convert_wt<<<(384*128 + 255) / 256, 256, 0, stream>>>((const float*)d_in[base + 0], lw + LW_EW1, 384, 128);
        convert_wt<<<(128*128 + 255) / 256, 256, 0, stream>>>((const float*)d_in[base + 1], lw + LW_EW2, 128, 128);
        convert_wt<<<(128*128 + 255) / 256, 256, 0, stream>>>((const float*)d_in[base + 2], lw + LW_EW3, 128, 128);
        convert_wt<<<(256*128 + 255) / 256, 256, 0, stream>>>((const float*)d_in[base + 8], lw + LW_NW1, 256, 128);
        convert_wt<<<(128*128 + 255) / 256, 256, 0, stream>>>((const float*)d_in[base + 9], lw + LW_NW2, 128, 128);
        convert_wt<<<(128*128 + 255) / 256, 256, 0, stream>>>((const float*)d_in[base +10], lw + LW_NW3, 128, 128);
    }

    for (int L = 0; L < NLAYERS; ++L) {
        int base = 3 + L * 16;
        unsigned short* lw = wt + (size_t)L * LW_SZ;

        edge_block<<<NE / 128, 256, 0, stream>>>(
            nA, eA, eidx,
            lw + LW_EW1, lw + LW_EW2, lw + LW_EW3,
            (const float*)d_in[base + 3], (const float*)d_in[base + 4], (const float*)d_in[base + 5],
            (const float*)d_in[base + 7] /*g*/, (const float*)d_in[base + 6] /*beta*/);

        hipMemsetAsync(agg, 0, (size_t)NN * D * sizeof(float), stream);
        scatter_add<<<(NE * 32) / 256, 256, 0, stream>>>(eA, eidx, agg);

        node_block<<<(NN + 127) / 128, 256, 0, stream>>>(
            nA, agg,
            lw + LW_NW1, lw + LW_NW2, lw + LW_NW3,
            (const float*)d_in[base + 11], (const float*)d_in[base + 12], (const float*)d_in[base + 13],
            (const float*)d_in[base + 15] /*g*/, (const float*)d_in[base + 14] /*beta*/);
    }

    hipMemcpyAsync(d_out, nA, (size_t)NN * D * sizeof(float), hipMemcpyDeviceToDevice, stream);
}
